// TargetsPredAttNet_45535243272620
// MI455X (gfx1250) — matrix operand order repaired and checked by reading, not compiled, not run on hardware
//
#include <hip/hip_runtime.h>

// ---------------------------------------------------------------------------
// TargetsPredAttNet for MI455X (gfx1250, wave32).
// Dense projections run on V_WMMA_F32_16X16X32_BF16 (bf16 in, f32 accum).
// Weights are pre-packed into a contiguous-k B-fragment layout (k=16h+s), staged once per
// block into LDS (<=40KB of the 320KB/WGP), and read as ds_load_b128 pairs.
// The GEMM is templated on K/32 (K is 128 or 160 only) and fully unrolled:
// all A fragments load up front in global clauses, then a branch-free chain
// of ds loads + 32/40 wmmas.  Edge softmax attention uses sortable-uint
// atomicMax + f32 atomicAdd scatter.  Workspace ~340 MB.  Param leaves
// assumed in JAX pytree order (sorted dict keys) -- see index table below.
// ---------------------------------------------------------------------------

static const int kB       = 1024;
static const int kNMAP    = 20000;
static const int kNCAND   = 200;
static const int kNPRED   = 30;
static const int kNT      = kB * kNPRED;   // 30720
static const int kE_NODE  = 320000;
static const int kE_NT    = 32768;
static const int kE_TRAJ  = kNT * 8;       // 245760
static const int kE_NTRAJ = kNT * 8;
static const int kCHUNK   = 51200;         // (B*NCAND)/4

// ---------------- device types ----------------
typedef __attribute__((ext_vector_type(16))) __bf16   v16bf;
typedef __attribute__((ext_vector_type(8)))  float    v8f;
typedef __attribute__((ext_vector_type(4)))  unsigned u32x4;

union FragBF { u32x4 q[2]; v16bf v; };

__device__ __forceinline__ unsigned short f2bf(float f) {
  unsigned u = __float_as_uint(f);
  unsigned r = ((u >> 16) & 1u) + 0x7fffu;   // round-to-nearest-even
  return (unsigned short)((u + r) >> 16);
}

// ---------------- utility kernels ----------------
__global__ __launch_bounds__(256) void k_fill_f32(float* p, float v, int n) {
  int t = blockIdx.x * 256 + threadIdx.x;
  if (t < n) p[t] = v;
}
__global__ __launch_bounds__(256) void k_fill_u32(unsigned* p, unsigned v, int n) {
  int t = blockIdx.x * 256 + threadIdx.x;
  if (t < n) p[t] = v;
}
__global__ __launch_bounds__(256) void k_cvt_bf16(const float* __restrict__ x,
                                                  unsigned short* __restrict__ y, int n) {
  int t = blockIdx.x * 256 + threadIdx.x;
  if (t < n) y[t] = f2bf(x[t]);
}

// Pack W[K,N] (f32 row-major, logical rows < Kreal, zero-padded to Kpad) into the
// 16x16x32 bf16 B-fragment layout.  u32 element t maps exactly to what k_gemm reads:
//   t = ((kt*NT16 + nt)*32 + lane)*8 + v ; value = { bf16 W[k0+1,n], bf16 W[k0,n] }
//   with n = nt*16 + (lane&15), k0 = kt*32 + (lane>>4)*16 + 2*v.
__global__ __launch_bounds__(256) void k_pack_w(const float* __restrict__ W,
                                                unsigned* __restrict__ out,
                                                int Kreal, int N, int total) {
  int t = blockIdx.x * 256 + threadIdx.x;
  if (t >= total) return;
  int NT16 = N >> 4;
  int kt = t / (NT16 * 256);
  int r  = t % (NT16 * 256);
  int nt = r >> 8;
  int r2 = r & 255;
  int lane = r2 >> 3;
  int v    = r2 & 7;
  int n  = nt * 16 + (lane & 15);
  int k0 = kt * 32 + ((lane >> 4) * 16) + 2 * v;
  unsigned short lo = (k0     < Kreal) ? f2bf(W[(size_t)k0 * N + n])       : (unsigned short)0;
  unsigned short hi = (k0 + 1 < Kreal) ? f2bf(W[(size_t)(k0 + 1) * N + n]) : (unsigned short)0;
  out[t] = ((unsigned)hi << 16) | (unsigned)lo;
}

// Y[M,N] = X[M,K] @ W[K,N] (+bias), K = KT*32 (KT = 4 or 5), fully unrolled.
// X bf16 row-major, W packed (above), Y f32.  block = 256 (8 waves); wave
// handles a 16-row strip x 8 n-tiles; grid.y = N/128.  The block's W slice
// (KT k-steps x 8 n-tiles x 1KB) is staged in LDS once.  M multiple of 16.
template <int KT>
__global__ __launch_bounds__(256) void k_gemm(const unsigned short* __restrict__ X,
                                              const unsigned* __restrict__ Wp,
                                              const float* __restrict__ bias,
                                              float* __restrict__ Y,
                                              int M, int N) {
  __shared__ u32x4 sW[KT * 512];             // KT * 8KB, <= 40KB
  const int wave = threadIdx.x >> 5;
  const int lane = threadIdx.x & 31;
  const int lrow = lane & 15;
  const int lhi  = lane >> 4;
  const int nblk = blockIdx.y * 8;
  const int NT16 = N >> 4;
  const int K    = KT * 32;
  const u32x4* W4 = (const u32x4*)Wp;

  // Cooperative copy of this block's weight slice into LDS (all threads,
  // BEFORE any early-out so the barrier is safe for partial blocks).
#pragma unroll
  for (int kt = 0; kt < KT; ++kt) {
    const u32x4* src = W4 + ((size_t)kt * NT16 + nblk) * 64;
#pragma unroll
    for (int j = 0; j < 2; ++j)
      sW[kt * 512 + j * 256 + threadIdx.x] = src[j * 256 + threadIdx.x];
  }
  __syncthreads();

  const int m0 = blockIdx.x * 128 + wave * 16;
  if (m0 >= M) return;                       // wave-uniform; EXEC stays all-ones

  v8f acc[8];
#pragma unroll
  for (int i = 0; i < 8; ++i)
#pragma unroll
    for (int j = 0; j < 8; ++j) acc[i][j] = 0.0f;

  const unsigned short* xrow = X + (size_t)(m0 + lrow) * K;
  const int koff = lhi * 16;

  // All A fragments up front (KT clauses of 2x global_load_b128 each).
  FragBF a[KT];
#pragma unroll
  for (int kt = 0; kt < KT; ++kt) {
    a[kt].q[0] = *(const u32x4*)(xrow + kt * 32 + koff);
    a[kt].q[1] = *(const u32x4*)(xrow + kt * 32 + 8 + koff);
  }
  if ((blockIdx.x + 1) * 128 + wave * 16 < M)
    __builtin_prefetch(xrow + (size_t)128 * K, 0, 0);   // next strip, global_prefetch_b8

  // Branch-free chain: KT * 8 wmmas fed by ds_load_b128 pairs.
#pragma unroll
  for (int kt = 0; kt < KT; ++kt) {
    const u32x4* wb = sW + kt * 512 + lane * 2;
#pragma unroll
    for (int nt = 0; nt < 8; ++nt) {
      FragBF b;
      b.q[0] = wb[nt * 64 + 0];
      b.q[1] = wb[nt * 64 + 1];
      acc[nt] = __builtin_amdgcn_wmma_f32_16x16x32_bf16(
          false, a[kt].v, false, b.v, (short)0, acc[nt], false, false);
    }
  }

#pragma unroll
  for (int nt = 0; nt < 8; ++nt) {
    const int ncol = (nblk + nt) * 16 + lrow;
    const float bv = bias ? bias[ncol] : 0.0f;
#pragma unroll
    for (int v = 0; v < 8; ++v)
      Y[(size_t)(m0 + lhi * 8 + v) * N + ncol] = acc[nt][v] + bv;
  }
}

// Row-wise GroupNorm over 128 features (one wave per row; 8 rows/block).
__global__ __launch_bounds__(256) void k_gn(const float* __restrict__ X,
                                            const float* __restrict__ g,
                                            const float* __restrict__ b,
                                            float* __restrict__ Y, int M, int relu) {
  const int wave = threadIdx.x >> 5, lane = threadIdx.x & 31;
  const int row = blockIdx.x * 8 + wave;
  if (row >= M) return;
  const float4 a = ((const float4*)(X + (size_t)row * 128))[lane];
  float s = a.x + a.y + a.z + a.w;
#pragma unroll
  for (int m = 16; m > 0; m >>= 1) s += __shfl_xor(s, m, 32);
  const float mu = s * (1.0f / 128.0f);
  const float dx = a.x - mu, dy = a.y - mu, dz = a.z - mu, dw = a.w - mu;
  float v = dx * dx + dy * dy + dz * dz + dw * dw;
#pragma unroll
  for (int m = 16; m > 0; m >>= 1) v += __shfl_xor(v, m, 32);
  const float inv = rsqrtf(v * (1.0f / 128.0f) + 1e-5f);
  const float4 gg = ((const float4*)g)[lane];
  const float4 bb = ((const float4*)b)[lane];
  float4 o;
  o.x = dx * inv * gg.x + bb.x;
  o.y = dy * inv * gg.y + bb.y;
  o.z = dz * inv * gg.z + bb.z;
  o.w = dw * inv * gg.w + bb.w;
  if (relu) { o.x = fmaxf(o.x, 0.f); o.y = fmaxf(o.y, 0.f); o.z = fmaxf(o.z, 0.f); o.w = fmaxf(o.w, 0.f); }
  ((float4*)(Y + (size_t)row * 128))[lane] = o;
}

__global__ __launch_bounds__(256) void k_add_relu(const float* __restrict__ a,
                                                  const float* __restrict__ b,
                                                  float* __restrict__ y, int n) {
  int t = blockIdx.x * 256 + threadIdx.x;
  if (t < n) y[t] = fmaxf(a[t] + b[t], 0.f);
}

// alpha[e,h] = dot(q[dst,h,:], k[src,h,:]) / sqrt(128);  one wave per edge.
__global__ __launch_bounds__(256) void k_edge_alpha(const float* __restrict__ q,
                                                    const float* __restrict__ k,
                                                    const int* __restrict__ src,
                                                    const int* __restrict__ dst,
                                                    float* __restrict__ alpha, int E) {
  const int wave = threadIdx.x >> 5, lane = threadIdx.x & 31;
  const int e = blockIdx.x * 8 + wave;
  if (e >= E) return;
  const int sn = src[e], dn = dst[e];
  const int h = lane >> 4, g = lane & 15;
  const float* qp = q + (size_t)dn * 256 + h * 128 + g * 8;
  const float* kp = k + (size_t)sn * 256 + h * 128 + g * 8;
  const float4 q0 = ((const float4*)qp)[0], q1 = ((const float4*)qp)[1];
  const float4 k0 = ((const float4*)kp)[0], k1 = ((const float4*)kp)[1];
  float p = q0.x * k0.x + q0.y * k0.y + q0.z * k0.z + q0.w * k0.w +
            q1.x * k1.x + q1.y * k1.y + q1.z * k1.z + q1.w * k1.w;
#pragma unroll
  for (int m = 8; m > 0; m >>= 1) p += __shfl_xor(p, m, 32);   // within 16-lane group
  if (g == 0) alpha[(size_t)e * 2 + h] = p * 0.08838834764831845f;  // 1/sqrt(128)
}

// segment max via sortable-uint atomicMax; one thread per (edge, head).
__global__ __launch_bounds__(256) void k_seg_max(const float* __restrict__ alpha,
                                                 const int* __restrict__ dst,
                                                 unsigned* __restrict__ m, int n2) {
  int t = blockIdx.x * 256 + threadIdx.x;
  if (t >= n2) return;
  int e = t >> 1, h = t & 1;
  unsigned u = __float_as_uint(alpha[t]);
  u = (u & 0x80000000u) ? ~u : (u | 0x80000000u);
  atomicMax(&m[(size_t)dst[e] * 2 + h], u);
}

__global__ __launch_bounds__(256) void k_exp_sum(float* __restrict__ alpha,
                                                 const int* __restrict__ dst,
                                                 const unsigned* __restrict__ m,
                                                 float* __restrict__ s, int n2) {
  int t = blockIdx.x * 256 + threadIdx.x;
  if (t >= n2) return;
  int e = t >> 1, h = t & 1;
  unsigned mu = m[(size_t)dst[e] * 2 + h];
  float mx = __uint_as_float((mu & 0x80000000u) ? (mu & 0x7fffffffu) : ~mu);
  float ev = __expf(alpha[t] - mx);
  alpha[t] = ev;
  atomicAdd(&s[(size_t)dst[e] * 2 + h], ev);
}

// acc[dst, :] += w * v[src, :]  (256 features = 2 heads x 128); one wave per edge.
__global__ __launch_bounds__(256) void k_edge_accum(const float* __restrict__ alpha,
                                                    const float* __restrict__ s,
                                                    const float* __restrict__ v,
                                                    const int* __restrict__ src,
                                                    const int* __restrict__ dst,
                                                    float* __restrict__ acc, int E) {
  const int wave = threadIdx.x >> 5, lane = threadIdx.x & 31;
  const int e = blockIdx.x * 8 + wave;
  if (e >= E) return;
  const int sn = src[e], dn = dst[e];
  const int h = lane >> 4;
  const float w = alpha[(size_t)e * 2 + h] / (s[(size_t)dn * 2 + h] + 1e-16f);
  const float* vp = v + (size_t)sn * 256 + lane * 8;
  float* ap = acc + (size_t)dn * 256 + lane * 8;
#pragma unroll
  for (int j = 0; j < 8; ++j) atomicAdd(ap + j, w * vp[j]);
}

// Y[n,c] += 0.5*(acc[n,c] + acc[n,128+c])   (mean over 2 heads, add skip result)
__global__ __launch_bounds__(256) void k_combine(float* __restrict__ Y,
                                                 const float* __restrict__ acc, int n) {
  int t = blockIdx.x * 256 + threadIdx.x;
  if (t >= n) return;
  int r = t >> 7, c = t & 127;
  Y[t] += 0.5f * (acc[(size_t)r * 256 + c] + acc[(size_t)r * 256 + 128 + c]);
}

// tc chunk (bf16, padded 160 cols): [targets(128) | cand0 | cand1 | zeros]
__global__ __launch_bounds__(256) void k_build_tc(const float* __restrict__ targets,
                                                  const float* __restrict__ cand,
                                                  unsigned short* __restrict__ tc,
                                                  int rowbase, int rows) {
  int t = blockIdx.x * 256 + threadIdx.x;
  if (t >= rows * 160) return;
  int r = t / 160, c = t % 160;
  int R = rowbase + r;
  int b = R / kNCAND, j = R % kNCAND;
  float val;
  if (c < 128)       val = targets[(size_t)b * 128 + c];
  else if (c == 128) val = cand[((size_t)b * kNCAND + j) * 2];
  else if (c == 129) val = cand[((size_t)b * kNCAND + j) * 2 + 1];
  else               val = 0.f;
  tc[t] = f2bf(val);
}

// out[row, j] = dot(Y[row,:128], W[:,j]) + bias[j], od in {1,2}; wave per row.
__global__ __launch_bounds__(256) void k_head(const float* __restrict__ Y,
                                              const float* __restrict__ W,
                                              const float* __restrict__ bias,
                                              float* __restrict__ out, int M, int od) {
  const int wave = threadIdx.x >> 5, lane = threadIdx.x & 31;
  const int row = blockIdx.x * 8 + wave;
  if (row >= M) return;
  const float4 a = ((const float4*)(Y + (size_t)row * 128))[lane];
  const int c = lane * 4;
  for (int j = 0; j < od; ++j) {
    float p = a.x * W[(c + 0) * od + j] + a.y * W[(c + 1) * od + j] +
              a.z * W[(c + 2) * od + j] + a.w * W[(c + 3) * od + j];
#pragma unroll
    for (int m = 16; m > 0; m >>= 1) p += __shfl_xor(p, m, 32);
    if (lane == 0) out[(size_t)row * od + j] = p + bias[j];
  }
}

__global__ __launch_bounds__(256) void k_argmax(const float* __restrict__ lg,
                                                int* __restrict__ col, int B) {
  const int wave = threadIdx.x >> 5, lane = threadIdx.x & 31;
  const int row = blockIdx.x * 8 + wave;
  if (row >= B) return;
  float best = -1e30f; int bi = 0;
  for (int i = lane; i < kNCAND; i += 32) {
    float v = lg[(size_t)row * kNCAND + i];
    if (v > best) { best = v; bi = i; }
  }
#pragma unroll
  for (int m = 16; m > 0; m >>= 1) {
    float ov = __shfl_xor(best, m, 32);
    int   oi = __shfl_xor(bi, m, 32);
    if (ov > best || (ov == best && oi < bi)) { best = ov; bi = oi; }
  }
  if (lane == 0) col[row] = bi;
}

__global__ __launch_bounds__(256) void k_softmax(const float* __restrict__ lg,
                                                 float* __restrict__ pr, int B) {
  const int wave = threadIdx.x >> 5, lane = threadIdx.x & 31;
  const int row = blockIdx.x * 8 + wave;
  if (row >= B) return;
  const float* x = lg + (size_t)row * kNCAND;
  float mx = -1e30f;
  for (int i = lane; i < kNCAND; i += 32) mx = fmaxf(mx, x[i]);
#pragma unroll
  for (int m = 16; m > 0; m >>= 1) mx = fmaxf(mx, __shfl_xor(mx, m, 32));
  float sum = 0.f;
  for (int i = lane; i < kNCAND; i += 32) sum += __expf(x[i] - mx);
#pragma unroll
  for (int m = 16; m > 0; m >>= 1) sum += __shfl_xor(sum, m, 32);
  const float inv = 1.0f / sum;
  for (int i = lane; i < kNCAND; i += 32)
    pr[(size_t)row * kNCAND + i] = __expf(x[i] - mx) * inv;
}

__global__ __launch_bounds__(256) void k_gather_target(const int* __restrict__ col,
                                                       const float* __restrict__ probs,
                                                       const float* __restrict__ cand,
                                                       const float* __restrict__ offs,
                                                       float* __restrict__ oprob,
                                                       float* __restrict__ opose,
                                                       float* __restrict__ tp, int B) {
  int b = blockIdx.x * 256 + threadIdx.x;
  if (b >= B) return;
  int c = col[b];
  oprob[b] = probs[(size_t)b * kNCAND + c];
  size_t i = ((size_t)b * kNCAND + c) * 2;
  float p0 = cand[i] + offs[i];
  float p1 = cand[i + 1] + offs[i + 1];
  opose[b * 2] = p0; opose[b * 2 + 1] = p1;
  tp[b * 2] = p0;    tp[b * 2 + 1] = p1;
}

__global__ __launch_bounds__(256) void k_frenet(const float* __restrict__ ip,
                                                const float* __restrict__ tp,
                                                float* __restrict__ p3, int n) {
  int i = blockIdx.x * 256 + threadIdx.x;
  if (i >= n) return;
  int b = i / kNPRED, ti = i % kNPRED;
  const float T = 0.1f * kNPRED;                 // 3.0
  float s = tp[b * 2], vs0 = ip[b * 3];
  float sa = 2.0f * (s - vs0 * T) / (T * T);
  float d = tp[b * 2 + 1], d0 = ip[b * 3 + 1], vd0 = ip[b * 3 + 2];
  float da = 2.0f * (d - d0 - vd0 * T) / (T * T);
  float t = (ti + 1) * 0.1f;
  float t2 = 0.5f * t * t;
  p3[(size_t)i * 3 + 0] = t;
  p3[(size_t)i * 3 + 1] = t * vs0 + t2 * sa;
  p3[(size_t)i * 3 + 2] = t * vd0 + t2 * da + d0;
}

__global__ __launch_bounds__(256) void k_spatial(const float* __restrict__ p3,
                                                 const float* __restrict__ w,
                                                 const float* __restrict__ b,
                                                 float* __restrict__ y, int nt) {
  int t = blockIdx.x * 256 + threadIdx.x;
  if (t >= nt * 128) return;
  int n = t >> 7, c = t & 127;
  y[t] = b[c] + p3[(size_t)n * 3] * w[c] + p3[(size_t)n * 3 + 1] * w[128 + c] +
         p3[(size_t)n * 3 + 2] * w[256 + c];
}

__global__ __launch_bounds__(256) void k_add_actor(const float* __restrict__ actors,
                                                   const int* __restrict__ ati,
                                                   const float* __restrict__ paths,
                                                   float* __restrict__ y, int nt) {
  int t = blockIdx.x * 256 + threadIdx.x;
  if (t >= nt * 128) return;
  int n = t >> 7, c = t & 127;
  y[t] = actors[(size_t)ati[n] * 128 + c] + paths[t];
}

// ---------------------------------------------------------------------------
extern "C" void kernel_launch(void* const* d_in, const int* in_sizes, int n_in,
                              void* d_out, int out_size, void* d_ws, size_t ws_size,
                              hipStream_t stream) {
  (void)in_sizes; (void)n_in; (void)out_size; (void)ws_size;

  // ---- raw inputs (setup_inputs insertion order) ----
  const float* actors      = (const float*)d_in[0];
  const float* nodes_in    = (const float*)d_in[1];
  const float* init_poses  = (const float*)d_in[2];
  // d_in[3] = target_poses_gt (unused by reference)
  const float* candidates  = (const float*)d_in[4];
  const int*   node_src    = (const int*)d_in[5];
  const int*   node_dst    = node_src + kE_NODE;
  const int*   nt_src      = (const int*)d_in[6];
  const int*   nt_dst      = nt_src + kE_NT;
  const int*   traj_src    = (const int*)d_in[7];
  const int*   traj_dst    = traj_src + kE_TRAJ;
  const int*   ntraj_src   = (const int*)d_in[8];
  const int*   ntraj_dst   = ntraj_src + kE_NTRAJ;
  const int*   actor_traj  = (const int*)d_in[9];

  // ---- params: JAX pytree order (sorted dict keys, recursively) ----
  // 10..18  h2o:  b, lr{b1,b2,g1,g2,w1,w2,wt}, w
  // 19..26  h2p:  b, lr{b1,b2,g1,g2,w1,w2}, w
  // 27..35  h2t:  b, lr{b1,b2,g1,g2,w1,w2,wt}, w
  // 36..51  node_blocks[0..1]: {bk,bq,bs,bv,wk,wq,ws,wv}
  // 52..67  path_blocks[0..1]
  // 68..69  spatial: b, w
  // 70..85  target_blocks[0..1]
  // 86..101 traj_blocks[0..1]
  auto F = [&](int i) { return (const float*)d_in[i]; };

  struct ConvParams { const float *bk,*bq,*bs,*bv,*wk,*wq,*ws,*wv; };
  auto conv_at = [&](int base) {
    ConvParams c;
    c.bk = F(base + 0); c.bq = F(base + 1); c.bs = F(base + 2); c.bv = F(base + 3);
    c.wk = F(base + 4); c.wq = F(base + 5); c.ws = F(base + 6); c.wv = F(base + 7);
    return c;
  };

  // ---- workspace bump allocator ----
  char* wsbase = (char*)d_ws;
  size_t wsoff = 0;
  auto alloc = [&](size_t bytes) -> void* {
    void* p = wsbase + wsoff;
    wsoff = (wsoff + bytes + 255) & ~(size_t)255;
    return p;
  };

  // ---- launch helpers ----
  auto fillf = [&](float* p, float v, int n) {
    k_fill_f32<<<(n + 255) / 256, 256, 0, stream>>>(p, v, n);
  };
  auto fillu = [&](unsigned* p, unsigned v, int n) {
    k_fill_u32<<<(n + 255) / 256, 256, 0, stream>>>(p, v, n);
  };
  auto cvt = [&](const float* x, unsigned short* y, int n) {
    k_cvt_bf16<<<(n + 255) / 256, 256, 0, stream>>>(x, y, n);
  };
  auto packW = [&](const float* W, int Kreal, int Kpad, int N) -> const unsigned* {
    unsigned* o = (unsigned*)alloc((size_t)Kpad * N * 2);
    int total = (Kpad / 32) * (N / 16) * 256;
    k_pack_w<<<(total + 255) / 256, 256, 0, stream>>>(W, o, Kreal, N, total);
    return o;
  };
  auto gemm = [&](const unsigned short* X, const unsigned* Wp, const float* bias,
                  float* Y, int M, int K, int N) {
    dim3 g((M + 127) / 128, N / 128);
    if (K == 128) k_gemm<4><<<g, 256, 0, stream>>>(X, Wp, bias, Y, M, N);
    else          k_gemm<5><<<g, 256, 0, stream>>>(X, Wp, bias, Y, M, N);
  };
  auto gn = [&](const float* X, const float* g, const float* b, float* Y, int M, int relu) {
    k_gn<<<(M + 7) / 8, 256, 0, stream>>>(X, g, b, Y, M, relu);
  };

  // ---- packed weights ----
  struct ConvPack { const unsigned *wq,*wk,*wv,*ws; const float *bq,*bk,*bv,*bs; };
  auto packConv = [&](const ConvParams& c) {
    ConvPack p;
    p.wq = packW(c.wq, 128, 128, 256);
    p.wk = packW(c.wk, 128, 128, 256);
    p.wv = packW(c.wv, 128, 128, 256);
    p.ws = packW(c.ws, 128, 128, 128);
    p.bq = c.bq; p.bk = c.bk; p.bv = c.bv; p.bs = c.bs;
    return p;
  };
  ConvPack pk_node0 = packConv(conv_at(36));
  ConvPack pk_node1 = packConv(conv_at(44));
  ConvPack pk_path0 = packConv(conv_at(52));
  ConvPack pk_path1 = packConv(conv_at(60));
  ConvPack pk_tgt0  = packConv(conv_at(70));
  ConvPack pk_tgt1  = packConv(conv_at(78));
  ConvPack pk_trj0  = packConv(conv_at(86));
  ConvPack pk_trj1  = packConv(conv_at(94));

  struct LrPack { const unsigned *w1,*w2,*wt; const float *g1,*b1,*g2,*b2; int Kpad; };
  LrPack pk_h2o = { packW(F(15), 130, 160, 128), packW(F(16), 128, 128, 128),
                    packW(F(17), 130, 160, 128), F(13), F(11), F(14), F(12), 160 };
  LrPack pk_h2t = { packW(F(32), 130, 160, 128), packW(F(33), 128, 128, 128),
                    packW(F(34), 130, 160, 128), F(30), F(28), F(31), F(29), 160 };
  LrPack pk_h2p = { packW(F(24), 128, 128, 128), packW(F(25), 128, 128, 128),
                    nullptr,                      F(22), F(20), F(23), F(21), 128 };
  const float* h2o_w = F(18); const float* h2o_b = F(10);
  const float* h2t_w = F(35); const float* h2t_b = F(27);
  const float* h2p_w = F(26); const float* h2p_b = F(19);
  const float* spat_b = F(68); const float* spat_w = F(69);

  // ---- big buffers ----
  const int kMaxNodes = (kNMAP > kNT) ? kNMAP : kNT;     // 30720
  float* nodesA   = (float*)alloc((size_t)kNMAP * 128 * 4);
  float* nodesB   = (float*)alloc((size_t)kNMAP * 128 * 4);
  float* targetsA = (float*)alloc((size_t)kB * 128 * 4);
  float* targetsB = (float*)alloc((size_t)kB * 128 * 4);
  float* pathsA   = (float*)alloc((size_t)kNT * 128 * 4);
  float* pathsB   = (float*)alloc((size_t)kNT * 128 * 4);
  float* trajsA   = (float*)alloc((size_t)kNT * 128 * 4);
  float* trajsB   = (float*)alloc((size_t)kNT * 128 * 4);
  unsigned short* bfA = (unsigned short*)alloc((size_t)kMaxNodes * 128 * 2);
  unsigned short* bfB = (unsigned short*)alloc((size_t)kMaxNodes * 128 * 2);
  float* qf   = (float*)alloc((size_t)kMaxNodes * 256 * 4);
  float* kf   = (float*)alloc((size_t)kMaxNodes * 256 * 4);
  float* vf   = (float*)alloc((size_t)kMaxNodes * 256 * 4);
  float* accf = (float*)alloc((size_t)kMaxNodes * 256 * 4);
  float* alphaf = (float*)alloc((size_t)kE_NODE * 2 * 4);
  unsigned* mb  = (unsigned*)alloc((size_t)kMaxNodes * 2 * 4);
  float* sb     = (float*)alloc((size_t)kMaxNodes * 2 * 4);
  float* L1f = (float*)alloc((size_t)kCHUNK * 128 * 4);
  float* L2f = (float*)alloc((size_t)kCHUNK * 128 * 4);
  float* LSf = (float*)alloc((size_t)kCHUNK * 128 * 4);
  unsigned short* L1b  = (unsigned short*)alloc((size_t)kCHUNK * 128 * 2);
  unsigned short* tcbf = (unsigned short*)alloc((size_t)kCHUNK * 160 * 2);
  int*   colbuf = (int*)alloc((size_t)kB * 4);
  float* tpbuf  = (float*)alloc((size_t)kB * 2 * 4);
  float* p3buf  = (float*)alloc((size_t)kNT * 3 * 4);

  // ---- output layout (reference return order, fp32, flat) ----
  float* out        = (float*)d_out;
  float* out_trajs  = out;            // 30720*2
  float* out_poses  = out + 61440;    // 1024*2
  float* out_prob   = out + 63488;    // 1024
  float* out_logits = out + 64512;    // 1024*200
  float* out_offs   = out + 269312;   // 1024*200*2
  float* out_probs  = out + 678912;   // 1024*200

  // ---- composite ops ----
  auto tconv = [&](const float* xsrc, int Ns, const float* xdst, int Nd,
                   const int* esrc, const int* edst, int E,
                   const ConvPack& W, float* Y) {
    cvt(xsrc, bfA, Ns * 128);
    const unsigned short* xdb;
    if (xdst == xsrc) { xdb = bfA; }
    else { cvt(xdst, bfB, Nd * 128); xdb = bfB; }
    gemm(xdb, W.wq, W.bq, qf, Nd, 128, 256);
    gemm(bfA, W.wk, W.bk, kf, Ns, 128, 256);
    gemm(bfA, W.wv, W.bv, vf, Ns, 128, 256);
    gemm(xdb, W.ws, W.bs, Y,  Nd, 128, 128);      // skip: x_dst @ ws + bs
    fillu(mb, 0x007fffffu, Nd * 2);               // sortable(-inf)
    fillf(sb, 0.f, Nd * 2);
    fillf(accf, 0.f, Nd * 256);
    k_edge_alpha<<<(E + 7) / 8, 256, 0, stream>>>(qf, kf, esrc, edst, alphaf, E);
    k_seg_max<<<(E * 2 + 255) / 256, 256, 0, stream>>>(alphaf, edst, mb, E * 2);
    k_exp_sum<<<(E * 2 + 255) / 256, 256, 0, stream>>>(alphaf, edst, mb, sb, E * 2);
    k_edge_accum<<<(E + 7) / 8, 256, 0, stream>>>(alphaf, sb, vf, esrc, edst, accf, E);
    k_combine<<<(Nd * 128 + 255) / 256, 256, 0, stream>>>(Y, accf, Nd * 128);
  };

  auto lin_res = [&](const unsigned short* Xb, const float* Xf_skip,
                     const LrPack& P, int M, float* OUT) {
    gemm(Xb, P.w1, nullptr, L1f, M, P.Kpad, 128);
    gn(L1f, P.g1, P.b1, L1f, M, 1);
    cvt(L1f, L1b, M * 128);
    gemm(L1b, P.w2, nullptr, L2f, M, 128, 128);
    gn(L2f, P.g2, P.b2, L2f, M, 0);
    const float* skip;
    if (P.wt) { gemm(Xb, P.wt, nullptr, LSf, M, P.Kpad, 128); skip = LSf; }
    else skip = Xf_skip;
    k_add_relu<<<(M * 128 + 255) / 256, 256, 0, stream>>>(L2f, skip, OUT, M * 128);
  };

  // ======================= pipeline =======================
  // 1) node blocks
  tconv(nodes_in, kNMAP, nodes_in, kNMAP, node_src, node_dst, kE_NODE, pk_node0, nodesB);
  tconv(nodesB,   kNMAP, nodesB,   kNMAP, node_src, node_dst, kE_NODE, pk_node1, nodesA);
  // 2) target blocks (src = final nodes)
  tconv(nodesA, kNMAP, actors,   kB, nt_src, nt_dst, kE_NT, pk_tgt0, targetsA);
  tconv(nodesA, kNMAP, targetsA, kB, nt_src, nt_dst, kE_NT, pk_tgt1, targetsB);
  // 3) candidate heads (chunked x4 over 204800 rows)
  for (int ch = 0; ch < 4; ++ch) {
    int rowbase = ch * kCHUNK;
    k_build_tc<<<(kCHUNK * 160 + 255) / 256, 256, 0, stream>>>(targetsB, candidates,
                                                               tcbf, rowbase, kCHUNK);
    lin_res(tcbf, nullptr, pk_h2t, kCHUNK, L1f);
    k_head<<<(kCHUNK + 7) / 8, 256, 0, stream>>>(L1f, h2t_w, h2t_b,
                                                 out_logits + rowbase, kCHUNK, 1);
    lin_res(tcbf, nullptr, pk_h2o, kCHUNK, L1f);
    k_head<<<(kCHUNK + 7) / 8, 256, 0, stream>>>(L1f, h2o_w, h2o_b,
                                                 out_offs + (size_t)rowbase * 2, kCHUNK, 2);
  }
  // 4) pick mode, softmax, gather pose
  k_argmax<<<(kB + 7) / 8, 256, 0, stream>>>(out_logits, colbuf, kB);
  k_softmax<<<(kB + 7) / 8, 256, 0, stream>>>(out_logits, out_probs, kB);
  k_gather_target<<<(kB + 255) / 256, 256, 0, stream>>>(colbuf, out_probs, candidates,
                                                        out_offs, out_prob, out_poses,
                                                        tpbuf, kB);
  // 5) frenet + spatial embed
  k_frenet<<<(kNT + 255) / 256, 256, 0, stream>>>(init_poses, tpbuf, p3buf, kNT);
  k_spatial<<<(kNT * 128 + 255) / 256, 256, 0, stream>>>(p3buf, spat_w, spat_b, pathsA, kNT);
  // 6) path blocks
  tconv(pathsA, kNT, pathsA, kNT, traj_src, traj_dst, kE_TRAJ, pk_path0, pathsB);
  tconv(pathsB, kNT, pathsB, kNT, traj_src, traj_dst, kE_TRAJ, pk_path1, pathsA);
  // 7) trajs = actors[idx] + paths, then traj blocks (src = final nodes)
  k_add_actor<<<(kNT * 128 + 255) / 256, 256, 0, stream>>>(actors, actor_traj, pathsA,
                                                           trajsA, kNT);
  tconv(nodesA, kNMAP, trajsA, kNT, ntraj_src, ntraj_dst, kE_NTRAJ, pk_trj0, trajsB);
  tconv(nodesA, kNMAP, trajsB, kNT, ntraj_src, ntraj_dst, kE_NTRAJ, pk_trj1, trajsA);
  // 8) final head -> trajs output
  cvt(trajsA, bfB, kNT * 128);
  lin_res(bfB, trajsA, pk_h2p, kNT, L1f);
  k_head<<<(kNT + 7) / 8, 256, 0, stream>>>(L1f, h2p_w, h2p_b, out_trajs, kNT, 2);
}
